// StaticWGAT_Layer_14508399525897
// MI455X (gfx1250) — matrix operand order repaired and checked by reading, not compiled, not run on hardware
//
#include <hip/hip_runtime.h>
#include <hip/hip_bf16.h>
#include <math.h>

#define NNODES 5000
#define NEDGES 80000
#define FEAT   2048   // 32 channels * 8 * 8
#define NODES_PER_BLOCK 8
#define CONV_BLOCKS (NNODES / NODES_PER_BLOCK)   // 625

typedef _Float16 v16h __attribute__((ext_vector_type(16)));
typedef _Float16 v8h  __attribute__((ext_vector_type(8)));
typedef float    v8f  __attribute__((ext_vector_type(8)));

__device__ __forceinline__ float gelu_exact(float x) {
  return 0.5f * x * (1.0f + erff(x * 0.70710678118654752f));
}

// ---------------------------------------------------------------------------
// Kernel 1: per-NODE conv via WMMA:  Y[n] = conv2d_3x3_same(h[n])  (no bias)
// Implicit GEMM: O[co, hw] = sum_{tap} W_tap[co, ci] * X_tap[ci, hw]
// 8 waves per block = 2 co-tiles x 4 hw-tiles; 9 K=32 WMMA steps per wave.
// Each block handles 8 consecutive nodes so the A-fragment (weight) loads and
// LDS zeroing are amortized 8x.
// ---------------------------------------------------------------------------
__global__ __launch_bounds__(256) void conv_wmma_kernel(
    const float* __restrict__ h,        // [N, 32, 8, 8]
    const float* __restrict__ conv_w,   // [32, 32, 3, 3]
    _Float16* __restrict__ Y)           // [N, 32, 64] f16
{
  // padded 10x10 spatial x 32 ci = 3200 halves; padded to 3584 (1792 dwords)
  // so the zero pass is a uniform 7 stores/thread (no divergence).
  __shared__ __align__(16) _Float16 Xs[3584];

  const int t      = threadIdx.x;
  const int lane   = t & 31;
  const int wave   = t >> 5;
  const int m_tile = wave >> 2;        // 0..1 (co tile)
  const int n_tile = wave & 3;         // 0..3 (hw tile)
  const int m      = lane & 15;
  const int hi     = (lane >> 4) & 1;  // lane half

  // zero LDS uniformly once (border stays zero across all node iterations:
  // interior stores never touch it)
  unsigned int* Xz = (unsigned int*)Xs;
#pragma unroll
  for (int i = 0; i < 7; ++i) Xz[t + i * 256] = 0u;

  // A fragments: contiguous K order k = 16*half + j (lane = M row), same order as B.
  // For each fragment half j, the 9 taps are 9 *consecutive* floats in conv_w
  // -> vectorizable global_load_b128s. Loaded once, reused for 8 nodes.
  v16h afrag[9];
  const int co = m_tile * 16 + m;
#pragma unroll
  for (int j = 0; j < 16; ++j) {
    const int v = j >> 1, p = j & 1;
    const int ci = hi * 16 + j; (void)v; (void)p;
    const float* wp = conv_w + co * 288 + ci * 9;
    float w9[9];
#pragma unroll
    for (int tap = 0; tap < 9; ++tap) w9[tap] = wp[tap];
#pragma unroll
    for (int tap = 0; tap < 9; ++tap) afrag[tap][j] = (_Float16)w9[tap];
  }

  // per-lane constants for the B fragment / output addressing
  const int n   = lane & 15;
  const int hw  = n_tile * 16 + n;
  const int y0  = hw >> 3, x0 = hw & 7;
  const int ci0 = hi * 16;

  const int node0 = blockIdx.x * NODES_PER_BLOCK;

  for (int it = 0; it < NODES_PER_BLOCK; ++it) {
    const int node = node0 + it;

    // stage h[node] into LDS: [pos(10x10)][ci] order, f16, interior only
    const float* hp = h + (size_t)node * FEAT;
    if (it + 1 < NODES_PER_BLOCK)                       // warm next node's row
      __builtin_prefetch(hp + FEAT + t * 4, 0, 1);
#pragma unroll
    for (int k = 0; k < 8; ++k) {
      const int idx = k * 256 + t;          // coalesced global read
      const int ci = idx >> 6;
      const int y  = (idx >> 3) & 7;
      const int x  = idx & 7;
      Xs[((y + 1) * 10 + (x + 1)) * 32 + ci] = (_Float16)hp[idx];
    }
    __syncthreads();

    // B layout: lanes 0-15 hold K=0-15 (ci0=0), lanes 16-31 K=16-31; N=lane%16
    v8f acc = {};
#pragma unroll
    for (int tap = 0; tap < 9; ++tap) {
      const int kh = tap / 3, kw = tap % 3;
      const int pos = (y0 + kh) * 10 + (x0 + kw);   // padded coords, in-range
      const _Float16* bp = &Xs[pos * 32 + ci0];
      const v8h blo = *(const v8h*)bp;              // ds_load_b128
      const v8h bhi8 = *(const v8h*)(bp + 8);       // ds_load_b128
      v16h b;
#pragma unroll
      for (int j = 0; j < 8; ++j) { b[j] = blo[j]; b[j + 8] = bhi8[j]; }
      acc = __builtin_amdgcn_wmma_f32_16x16x32_f16(false, afrag[tap], false, b,
                                                   (short)0, acc, false, false);
    }

    // C/D layout: VGPR r -> M=r (lanes 0-15) / M=r+8 (lanes 16-31); N=lane%16
    _Float16* yp = Y + (size_t)node * FEAT;
#pragma unroll
    for (int r = 0; r < 8; ++r) {
      const int coo = m_tile * 16 + r + hi * 8;
      yp[coo * 64 + hw] = (_Float16)acc[r];
    }
    __syncthreads();   // all waves done reading Xs before next restage
  }
}

// ---------------------------------------------------------------------------
// CSR build: zero counts -> histogram(dst) -> scan -> scatter edge ids
// ---------------------------------------------------------------------------
__global__ void zero_kernel(int* __restrict__ p, int n) {
  const int i = blockIdx.x * blockDim.x + threadIdx.x;
  if (i < n) p[i] = 0;
}

__global__ void hist_kernel(const int* __restrict__ dst, int* __restrict__ counts, int n) {
  const int i = blockIdx.x * blockDim.x + threadIdx.x;
  if (i < n) atomicAdd(&counts[dst[i]], 1);
}

__global__ __launch_bounds__(256) void scan_kernel(
    const int* __restrict__ counts, int* __restrict__ offsets,
    int* __restrict__ cursor, int n)
{
  __shared__ int partial[256];
  const int t = threadIdx.x;
  const int chunk = (n + 255) / 256;
  const int base = t * chunk;
  int s = 0;
  for (int i = 0; i < chunk; ++i)
    if (base + i < n) s += counts[base + i];
  partial[t] = s;
  __syncthreads();
  if (t == 0) {
    int run = 0;
    for (int i = 0; i < 256; ++i) { int v = partial[i]; partial[i] = run; run += v; }
    offsets[n] = run;
  }
  __syncthreads();
  int run = partial[t];
  for (int i = 0; i < chunk; ++i) {
    if (base + i < n) {
      offsets[base + i] = run;
      cursor[base + i]  = run;
      run += counts[base + i];
    }
  }
}

__global__ void scatter_kernel(const int* __restrict__ dst, int* __restrict__ cursor,
                               int* __restrict__ edge_ids, int n) {
  const int i = blockIdx.x * blockDim.x + threadIdx.x;
  if (i < n) {
    const int pos = atomicAdd(&cursor[dst[i]], 1);
    edge_ids[pos] = i;
  }
}

// ---------------------------------------------------------------------------
// Kernel 6: per-dst-node gather:
//   out[n] = max_{edges j->n} gelu(e_j * Y[src_j] + b)
// Exact GELU is unimodal (single min at x ~= -0.7518), so
//   max_j gelu(x_j) == max(gelu(max_j x_j), gelu(min_j x_j))
// -> edge loop tracks only min/max of the pre-activation (1 FMA + max + min
// per element), and erf runs twice per output element instead of per edge.
// One block per node; each thread owns 8 contiguous elements (b128 Y loads).
// max/min are exactly commutative -> bit-deterministic despite scatter order.
// ---------------------------------------------------------------------------
__global__ __launch_bounds__(256) void gather_max_kernel(
    const int* __restrict__ offsets, const int* __restrict__ edge_ids,
    const int* __restrict__ src, const float* __restrict__ ew,
    const float* __restrict__ conv_b, const _Float16* __restrict__ Y,
    float* __restrict__ out)
{
  const int node = blockIdx.x;
  const int t = threadIdx.x;
  const int beg = offsets[node];
  const int end = offsets[node + 1];
  const int e0 = t * 8;                      // 8 contiguous elements per thread

  float bias[8], xmax[8], xmin[8];
#pragma unroll
  for (int i = 0; i < 8; ++i) {
    bias[i] = conv_b[(e0 + i) >> 6];
    xmax[i] = -INFINITY;
    xmin[i] = INFINITY;
  }

  for (int j = beg; j < end; ++j) {
    const int eid = edge_ids[j];
    const float w = ew[eid];
    const v8h ys = *(const v8h*)(Y + (size_t)src[eid] * FEAT + e0);

    // prefetch next edge's Y row into cache while we compute
    const int jn = (j + 1 < end) ? (j + 1) : j;
    __builtin_prefetch(Y + (size_t)src[edge_ids[jn]] * FEAT + e0, 0, 1);

#pragma unroll
    for (int i = 0; i < 8; ++i) {
      const float x = fmaf(w, (float)ys[i], bias[i]);
      xmax[i] = fmaxf(xmax[i], x);
      xmin[i] = fminf(xmin[i], x);
    }
  }

  float r[8];
#pragma unroll
  for (int i = 0; i < 8; ++i)
    r[i] = fmaxf(gelu_exact(xmax[i]), gelu_exact(xmin[i]));

  float4* op = (float4*)(out + (size_t)node * FEAT + e0);
  op[0] = make_float4(r[0], r[1], r[2], r[3]);
  op[1] = make_float4(r[4], r[5], r[6], r[7]);
}

// ---------------------------------------------------------------------------
extern "C" void kernel_launch(void* const* d_in, const int* in_sizes, int n_in,
                              void* d_out, int out_size, void* d_ws, size_t ws_size,
                              hipStream_t stream) {
  // setup_inputs() dict order: h, e, conv_w, conv_b, src, dst
  const float* h      = (const float*)d_in[0];
  const float* ew     = (const float*)d_in[1];
  const float* conv_w = (const float*)d_in[2];
  const float* conv_b = (const float*)d_in[3];
  const int*   src    = (const int*)d_in[4];
  const int*   dst    = (const int*)d_in[5];
  float* out = (float*)d_out;

  // workspace layout
  char* ws = (char*)d_ws;
  _Float16* Y   = (_Float16*)ws;                                  // 20,480,000 B
  int* counts   = (int*)(ws + (size_t)NNODES * FEAT * 2);
  int* offsets  = counts + NNODES;
  int* cursor   = offsets + (NNODES + 1);
  int* edge_ids = cursor + NNODES;

  conv_wmma_kernel<<<CONV_BLOCKS, 256, 0, stream>>>(h, conv_w, Y);
  zero_kernel<<<(NNODES + 255) / 256, 256, 0, stream>>>(counts, NNODES);
  hist_kernel<<<(NEDGES + 255) / 256, 256, 0, stream>>>(dst, counts, NEDGES);
  scan_kernel<<<1, 256, 0, stream>>>(counts, offsets, cursor, NNODES);
  scatter_kernel<<<(NEDGES + 255) / 256, 256, 0, stream>>>(dst, cursor, edge_ids, NEDGES);
  gather_max_kernel<<<NNODES, 256, 0, stream>>>(offsets, edge_ids, src, ew, conv_b, Y, out);
}
